// prompt_model_31112743092275
// MI455X (gfx1250) — matrix operand order repaired and checked by reading, not compiled, not run on hardware
//
#include <hip/hip_runtime.h>
#include <hip/hip_bf16.h>

#define BB 4
#define SS 512
#define HH 768
#define VV 30522
#define AA 200
#define MM (BB*SS)      // 2048 tokens
#define NPAD 30720      // vocab rows padded to multiple of 256 (macro-tile width)

typedef __attribute__((ext_vector_type(16))) __bf16 v16bf;
typedef __attribute__((ext_vector_type(8)))  float  v8f;

union P8 { uint4 u; __bf16 h[8]; };

__device__ __forceinline__ __bf16 f2bf(float x) {
  unsigned int u = __builtin_bit_cast(unsigned int, x);
  u += 0x7FFFu + ((u >> 16) & 1u);          // round-to-nearest-even
  unsigned short hs = (unsigned short)(u >> 16);
  return __builtin_bit_cast(__bf16, hs);
}

// ---------------------------------------------------------------------------
// K0: fp32 -> bf16 conversion (zero-fills [ncopy, ntotal) padding region)
// ---------------------------------------------------------------------------
__global__ __launch_bounds__(256)
void k_f32_to_bf16(const float* __restrict__ src, __bf16* __restrict__ dst,
                   long ncopy, long ntotal) {
  long i = (long)blockIdx.x * 256 + threadIdx.x;
  if (i < ntotal) dst[i] = (i < ncopy) ? f2bf(src[i]) : f2bf(0.f);
}

// ---------------------------------------------------------------------------
// K1: s[t] = tanh(hidden[t]·W1^T + b1) · v_attn   (8 tokens per block)
// ---------------------------------------------------------------------------
__global__ __launch_bounds__(256)
void k_attn_score(const float* __restrict__ hidden, const float* __restrict__ W_first,
                  const float* __restrict__ b_first, const float* __restrict__ v_attn,
                  float* __restrict__ s_out) {
  __shared__ float hsh[8 * HH];
  __shared__ float red[256];
  const int tid = threadIdx.x;
  const int t0  = blockIdx.x * 8;
  for (int i = tid; i < 8 * HH; i += 256) hsh[i] = hidden[(size_t)t0 * HH + i];
  __syncthreads();

  float val[8];
#pragma unroll
  for (int j = 0; j < 8; ++j) val[j] = 0.f;

  if (tid < AA) {
    float acc[8];
    const float b0 = b_first[tid];
#pragma unroll
    for (int j = 0; j < 8; ++j) acc[j] = b0;
    const float* wr = W_first + (size_t)tid * HH;
    for (int h = 0; h < HH; ++h) {
      const float wv = wr[h];
#pragma unroll
      for (int j = 0; j < 8; ++j) acc[j] = fmaf(wv, hsh[j * HH + h], acc[j]);
    }
    const float va = v_attn[tid];
#pragma unroll
    for (int j = 0; j < 8; ++j) val[j] = tanhf(acc[j]) * va;
  }

  for (int j = 0; j < 8; ++j) {
    red[tid] = val[j];
    __syncthreads();
    for (int st = 128; st > 0; st >>= 1) {
      if (tid < st) red[tid] += red[tid + st];
      __syncthreads();
    }
    if (tid == 0) s_out[t0 + j] = red[0];
    __syncthreads();
  }
}

// ---------------------------------------------------------------------------
// K2: per batch: max-reduce, e = exp(s - max), inclusive scan, suffix, total
// ---------------------------------------------------------------------------
__global__ __launch_bounds__(SS)
void k_softmax_prep(const float* __restrict__ s, float* __restrict__ e_out,
                    float* __restrict__ cum_out, float* __restrict__ suf_out,
                    float* __restrict__ tot_out) {
  __shared__ float sh[SS];
  const int b = blockIdx.x;
  const int k = threadIdx.x;
  const float sv = s[b * SS + k];
  sh[k] = sv; __syncthreads();
  for (int st = SS / 2; st > 0; st >>= 1) {
    if (k < st) sh[k] = fmaxf(sh[k], sh[k + st]);
    __syncthreads();
  }
  const float m = sh[0];
  __syncthreads();
  const float e = expf(sv - m);
  e_out[b * SS + k] = e;
  sh[k] = e; __syncthreads();
  // Hillis-Steele inclusive scan
  for (int off = 1; off < SS; off <<= 1) {
    const float add = (k >= off) ? sh[k - off] : 0.f;
    __syncthreads();
    sh[k] += add;
    __syncthreads();
  }
  const float cum = sh[k];
  const float tot = sh[SS - 1];
  cum_out[b * SS + k] = cum;
  suf_out[b * SS + k] = tot - cum + e;   // sum over k' >= k
  if (k == 0) tot_out[b] = tot;
}

// ---------------------------------------------------------------------------
// K3: cumulative-scan attention -> hidd (fp32 to d_out) + bf16 copy for GEMM
// type 0: prefix (k<=j), 1: suffix (k>=j), 2: leave-one-out (k!=j)
// ---------------------------------------------------------------------------
__global__ __launch_bounds__(HH)
void k_context(const float* __restrict__ hidden, const int* __restrict__ ptype,
               const float* __restrict__ e, const float* __restrict__ cum,
               const float* __restrict__ suf, const float* __restrict__ tot,
               float* __restrict__ hidd_out, __bf16* __restrict__ hidd_bf) {
  __shared__ float esh[SS];
  __shared__ float csh[SS];
  const int b = blockIdx.x;
  const int h = threadIdx.x;
  const int type = ptype[b];
  for (int i = h; i < SS; i += HH) {
    esh[i] = e[b * SS + i];
    csh[i] = (type == 0) ? cum[b * SS + i] : ((type == 1) ? suf[b * SS + i] : 0.f);
  }
  __syncthreads();

  const size_t base = (size_t)b * SS * HH + h;
  if (type == 0) {
    float run = 0.f;
    for (int k = 0; k < SS; ++k) {
      const size_t idx = base + (size_t)k * HH;
      run = fmaf(esh[k], hidden[idx], run);
      const float o = run / (csh[k] * (float)(k + 1));
      hidd_out[idx] = o;
      hidd_bf[idx]  = f2bf(o);
    }
  } else if (type == 1) {
    float run = 0.f;
    for (int k = SS - 1; k >= 0; --k) {
      const size_t idx = base + (size_t)k * HH;
      run = fmaf(esh[k], hidden[idx], run);
      const float o = run / (csh[k] * (float)(SS - k));
      hidd_out[idx] = o;
      hidd_bf[idx]  = f2bf(o);
    }
  } else {
    const float totv = tot[b];
    float th = 0.f;
    for (int k = 0; k < SS; ++k) th = fmaf(esh[k], hidden[base + (size_t)k * HH], th);
    const float inv = 1.f / (float)(SS - 1);
    for (int k = 0; k < SS; ++k) {
      const size_t idx = base + (size_t)k * HH;
      const float hv = hidden[idx];
      const float o  = (th - esh[k] * hv) / (totv - esh[k]) * inv;
      hidd_out[idx] = o;
      hidd_bf[idx]  = f2bf(o);
    }
  }
}

// ---------------------------------------------------------------------------
// WMMA GEMM: C[M,Nreal] = A[M,K](bf16) * Bw[N,K]^T(bf16) + bias
// 256 threads = 8 waves (2x4); each wave computes a 32x64 tile (2x4 WMMA tiles)
// ---------------------------------------------------------------------------
__global__ __launch_bounds__(256)
void k_gemm_wmma(const __bf16* __restrict__ A, const __bf16* __restrict__ Bw,
                 float* __restrict__ C, const float* __restrict__ bias,
                 int Kd, int ldc, int Nreal) {
  const int tid  = threadIdx.x;
  const int lane = tid & 31;
  const int w    = tid >> 5;
  const int wm   = w >> 2, wn = w & 3;
  const int lm   = lane & 15;
  const int cc   = lane >> 4;             // lane half: 0 or 1
  const int row0 = blockIdx.y * 64 + wm * 32;
  const int col0 = blockIdx.x * 256 + wn * 64;

  v8f acc[2][4];
#pragma unroll
  for (int mt = 0; mt < 2; ++mt)
#pragma unroll
    for (int nt = 0; nt < 4; ++nt)
      acc[mt][nt] = (v8f){0.f, 0.f, 0.f, 0.f, 0.f, 0.f, 0.f, 0.f};

  for (int k0 = 0; k0 < Kd; k0 += 32) {
    // A fragment, 16x32 bf16: lane half cc, elem e -> k_local = 8*cc + e (+8 for e>=8)
    v16bf af[2];
#pragma unroll
    for (int mt = 0; mt < 2; ++mt) {
      const __bf16* pa = A + (size_t)(row0 + mt * 16 + lm) * Kd + k0 + 8 * cc;
      P8 lo, hi;
      lo.u = *(const uint4*)(pa);
      hi.u = *(const uint4*)(pa + 16);
      v16bf f;
#pragma unroll
      for (int e = 0; e < 8; ++e) { f[e] = lo.h[e]; f[e + 8] = hi.h[e]; }
      af[mt] = f;
    }
    // B fragment, 32x16 bf16: lane half cc, elem e -> k_local = 8*cc + e (+8 for e>=8)
    v16bf bfv[4];
#pragma unroll
    for (int nt = 0; nt < 4; ++nt) {
      const __bf16* pb = Bw + (size_t)(col0 + nt * 16 + lm) * Kd + k0 + 8 * cc;
      P8 lo, hi;
      lo.u = *(const uint4*)(pb);
      hi.u = *(const uint4*)(pb + 16);
      v16bf f;
#pragma unroll
      for (int e = 0; e < 8; ++e) { f[e] = lo.h[e]; f[e + 8] = hi.h[e]; }
      bfv[nt] = f;
      __builtin_prefetch((const void*)(pb + 64), 0, 1);  // k-block two steps ahead in this row
    }
#pragma unroll
    for (int mt = 0; mt < 2; ++mt)
#pragma unroll
      for (int nt = 0; nt < 4; ++nt)
        acc[mt][nt] = __builtin_amdgcn_wmma_f32_16x16x32_bf16(
            false, af[mt], false, bfv[nt], (short)0, acc[mt][nt], false, false);
  }

  // D layout: VGPR r, lane -> D[M = r + 8*cc][N = lane%16]
#pragma unroll
  for (int mt = 0; mt < 2; ++mt)
#pragma unroll
    for (int nt = 0; nt < 4; ++nt) {
      const int col = col0 + nt * 16 + lm;
      if (col < Nreal) {
        const float bv = bias ? bias[col] : 0.f;
#pragma unroll
        for (int r = 0; r < 8; ++r) {
          const int row = row0 + mt * 16 + 8 * cc + r;
          C[(size_t)row * ldc + col] = acc[mt][nt][r] + bv;
        }
      }
    }
}

// ---------------------------------------------------------------------------
// K4: exact GELU + LayerNorm(eps=1e-12) per token -> bf16 xn
// ---------------------------------------------------------------------------
__global__ __launch_bounds__(256)
void k_gelu_ln(const float* __restrict__ x, const float* __restrict__ gamma,
               const float* __restrict__ beta, __bf16* __restrict__ xn_bf) {
  __shared__ float red[256];
  const int t   = blockIdx.x;
  const int tid = threadIdx.x;
  const float* xr = x + (size_t)t * HH;

  float g[3];
  float ps = 0.f;
#pragma unroll
  for (int i = 0; i < 3; ++i) {
    const float v  = xr[tid + i * 256];
    const float gv = 0.5f * v * (1.f + erff(v * 0.70710678118654752f));
    g[i] = gv;
    ps += gv;
  }
  red[tid] = ps; __syncthreads();
  for (int st = 128; st > 0; st >>= 1) {
    if (tid < st) red[tid] += red[tid + st];
    __syncthreads();
  }
  const float mu = red[0] / (float)HH;
  __syncthreads();

  float pv = 0.f;
#pragma unroll
  for (int i = 0; i < 3; ++i) { const float d = g[i] - mu; pv += d * d; }
  red[tid] = pv; __syncthreads();
  for (int st = 128; st > 0; st >>= 1) {
    if (tid < st) red[tid] += red[tid + st];
    __syncthreads();
  }
  const float var = red[0] / (float)HH;
  const float inv = rsqrtf(var + 1e-12f);
#pragma unroll
  for (int i = 0; i < 3; ++i) {
    const int c = tid + i * 256;
    const float v = (g[i] - mu) * inv * gamma[c] + beta[c];
    xn_bf[(size_t)t * HH + c] = f2bf(v);
  }
}

// ---------------------------------------------------------------------------
extern "C" void kernel_launch(void* const* d_in, const int* in_sizes, int n_in,
                              void* d_out, int out_size, void* d_ws, size_t ws_size,
                              hipStream_t stream) {
  (void)in_sizes; (void)n_in; (void)out_size; (void)ws_size;

  const float* hidden  = (const float*)d_in[0];
  const int*   ptype   = (const int*)  d_in[1];
  const float* W_first = (const float*)d_in[2];
  const float* b_first = (const float*)d_in[3];
  const float* v_attn  = (const float*)d_in[4];
  const float* W_fc    = (const float*)d_in[5];
  const float* b_fc    = (const float*)d_in[6];
  const float* ln_g    = (const float*)d_in[7];
  const float* ln_b    = (const float*)d_in[8];
  const float* W_out   = (const float*)d_in[9];
  const float* b_out   = (const float*)d_in[10];

  float* out_hidd   = (float*)d_out;                      // [B,S,H]
  float* out_logits = out_hidd + (size_t)MM * HH;         // [B,S,V]

  // workspace carve-out (~61 MB)
  char*  wsp = (char*)d_ws;
  size_t off = 0;
  auto carve = [&](size_t bytes) -> char* {
    char* p = wsp + off;
    off += (bytes + 255) & ~(size_t)255;
    return p;
  };
  __bf16* wout_bf = (__bf16*)carve((size_t)NPAD * HH * 2);
  __bf16* wfc_bf  = (__bf16*)carve((size_t)HH * HH * 2);
  __bf16* hidd_bf = (__bf16*)carve((size_t)MM * HH * 2);
  __bf16* xn_bf   = (__bf16*)carve((size_t)MM * HH * 2);
  float*  x_f32   = (float*) carve((size_t)MM * HH * 4);
  float*  s_f     = (float*) carve((size_t)MM * 4);
  float*  e_f     = (float*) carve((size_t)MM * 4);
  float*  cum_f   = (float*) carve((size_t)MM * 4);
  float*  suf_f   = (float*) carve((size_t)MM * 4);
  float*  tot_f   = (float*) carve((size_t)BB * 4);

  // K0: weight conversions (W_out padded+zero-filled to NPAD rows)
  {
    const long ntot = (long)NPAD * HH, ncpy = (long)VV * HH;
    k_f32_to_bf16<<<dim3((unsigned)((ntot + 255) / 256)), dim3(256), 0, stream>>>(
        W_out, wout_bf, ncpy, ntot);
  }
  {
    const long n = (long)HH * HH;
    k_f32_to_bf16<<<dim3((unsigned)((n + 255) / 256)), dim3(256), 0, stream>>>(
        W_fc, wfc_bf, n, n);
  }

  // K1: attention scores
  k_attn_score<<<dim3(MM / 8), dim3(256), 0, stream>>>(hidden, W_first, b_first, v_attn, s_f);

  // K2: softmax precompute (max, exp, prefix/suffix/total)
  k_softmax_prep<<<dim3(BB), dim3(SS), 0, stream>>>(s_f, e_f, cum_f, suf_f, tot_f);

  // K3: scan-based masked attention -> hidd (output 1) + bf16 copy
  k_context<<<dim3(BB), dim3(HH), 0, stream>>>(hidden, ptype, e_f, cum_f, suf_f, tot_f,
                                               out_hidd, hidd_bf);

  // GEMM1: x = hidd @ W_fc^T + b_fc  (2048 x 768 x 768)
  k_gemm_wmma<<<dim3(HH / 256, MM / 64), dim3(256), 0, stream>>>(
      hidd_bf, wfc_bf, x_f32, b_fc, HH, HH, HH);

  // K4: GELU + LayerNorm -> xn (bf16)
  k_gelu_ln<<<dim3(MM), dim3(256), 0, stream>>>(x_f32, ln_g, ln_b, xn_bf);

  // GEMM2: logits = xn @ W_out^T + b_out  (2048 x 30522 x 768)
  k_gemm_wmma<<<dim3(NPAD / 256, MM / 64), dim3(256), 0, stream>>>(
      xn_bf, wout_bf, out_logits, b_out, HH, VV, VV);
}
